// Model_70695161692795
// MI455X (gfx1250) — matrix operand order repaired and checked by reading, not compiled, not run on hardware
//
#include <hip/hip_runtime.h>
#include <hip/hip_bf16.h>

// Fused 2-layer LSTM (B=2048, T=512, F=16, H=50) + FC + sigmoid for gfx1250.
// One workgroup (256 thr = 8 waves) owns a 16-row batch tile and iterates all
// 512 timesteps, keeping h/c state of both layers in LDS and all weight
// fragments in VGPRs. Matrix work uses v_wmma_f32_16x16x32_bf16.
// Hidden state is stored in LDS in WMMA A-fragment order so the matrix phase
// loads each A fragment as one contiguous v16bf (2x ds_load_b128).
// Activations use native v_tanh_f32 (sigmoid = 0.5*tanh(x/2)+0.5).
// Padding: H=50 -> 64, gates 200 -> 256 (4 blocks of 64). Padded region stays 0.

typedef __attribute__((ext_vector_type(16))) __bf16 v16bf;
typedef __attribute__((ext_vector_type(8)))  float  v8f;

#define WMMA_BF16(a, b, c) \
  __builtin_amdgcn_wmma_f32_16x16x32_bf16(false, (a), false, (b), (short)0, (c), false, false)

// ---------- fast branchless activations (native v_tanh / v_exp / v_rcp) ----
__device__ __forceinline__ float fast_rcp(float x) {
#if __has_builtin(__builtin_amdgcn_rcpf)
  return __builtin_amdgcn_rcpf(x);
#else
  return 1.0f / x;
#endif
}
__device__ __forceinline__ float tanh_fast(float x) {
#if __has_builtin(__builtin_amdgcn_tanhf)
  return __builtin_amdgcn_tanhf(x);
#elif __has_builtin(__builtin_amdgcn_tanh_f32)
  return __builtin_amdgcn_tanh_f32(x);
#else
  // tanh(x) = 1 - 2/(1+exp(2x)); exp->inf / ->0 saturates correctly to +/-1.
  return 1.0f - 2.0f * fast_rcp(1.0f + __expf(2.0f * x));
#endif
}
__device__ __forceinline__ float sigf(float x) {
  // sigmoid(x) = 0.5*tanh(x/2) + 0.5 : single TRANS op on gfx1250.
  return fmaf(0.5f, tanh_fast(0.5f * x), 0.5f);
}

// A-matrix (16x32 bf16) per-lane K mapping (ISA 7.12.2, 16-bit A 16x32):
// lanes 0-15: V0-3 -> K0..7, V4-7 -> K16..23 ; lanes16-31: K8..15, K24..31
// a_k(e,kh) = e<8 ? 8*kh+e : 8+8*kh+e.  Inverse (k' in [0,32)):
//   k'<16: kh=k'>>3,        e=k'&7
//   else : kh=(k'&15)>>3,   e=(k'&7)+8
__device__ __forceinline__ void inv_ak(int kp, int& kh, int& e) {
  if (kp < 16) { kh = kp >> 3;        e = kp & 7; }
  else         { kh = (kp & 15) >> 3; e = (kp & 7) + 8; }
}

// Build B-fragment (K x 16) of W^T where W is row-major [200][ldw] (real dims).
// B layout (matches A): lanes 0-15 hold K=kbase+0..7,16..23,
// lanes 16-31 hold K=kbase+8..15,24..31; n = lane&15 within tile at n0.
__device__ __forceinline__ v16bf load_wfragT(const float* W, int ldw, int kreal,
                                             int n0, int lane, int kbase) {
  int nn = n0 + (lane & 15);
  int kh = lane >> 4;
  int gate = nn >> 6, j = nn & 63;
  bool nv = (j < 50);
  int nreal = gate * 50 + j;
  v16bf w;
#pragma unroll
  for (int e = 0; e < 16; e++) {
    int k = kbase + (e < 8 ? kh * 8 + e : 16 + kh * 8 + (e - 8));
    float v = (nv && k < kreal) ? W[nreal * ldw + k] : 0.0f;
    w[e] = (__bf16)v;
  }
  return w;
}

__global__ __launch_bounds__(256, 1) void lstm2_fused_kernel(
    const float* __restrict__ x,    const float* __restrict__ Wih0,
    const float* __restrict__ Whh0, const float* __restrict__ b0,
    const float* __restrict__ Wih1, const float* __restrict__ Whh1,
    const float* __restrict__ b1,   const float* __restrict__ Wfc,
    const float* __restrict__ bfc,  float* __restrict__ out) {
  constexpr int T = 512, F = 16;

  // A-fragment-ordered LDS state: [chunk][lane][elem], 32B contiguous per lane.
  __shared__ __align__(32) __bf16 xsf[32][16];         // x_t fragment (K=0..31)
  __shared__ __align__(32) __bf16 h0frag[2][32][16];   // layer-0 hidden
  __shared__ __align__(32) __bf16 h1frag[2][32][16];   // layer-1 hidden
  __shared__ float c0s[16][64];                        // cell states (f32)
  __shared__ float c1s[16][64];
  __shared__ float h1last[16][64];                     // f32 h1 at t=T-1 (FC)
  __shared__ float gatesS[16][260];                    // raw gates [16][256]+pad

  const int tid  = threadIdx.x;
  const int lane = tid & 31, wave = tid >> 5;
  const int ln = lane & 15, khl = lane >> 4;
  const int b0r = blockIdx.x * 16;

  // ---- init state to zero (incl. the constant-zero K>=16 half of xsf) ----
#pragma unroll
  for (int u = 0; u < 2; u++) {
    int idx = tid + 256 * u;          // 512 = 32*16 elements
    ((__bf16*)xsf)[idx] = (__bf16)0.0f;
  }
#pragma unroll
  for (int u = 0; u < 4; u++) {
    int idx = tid + 256 * u;          // 1024 elements each
    ((__bf16*)h0frag)[idx] = (__bf16)0.0f;
    ((__bf16*)h1frag)[idx] = (__bf16)0.0f;
    ((float*)c0s)[idx] = 0.0f;
    ((float*)c1s)[idx] = 0.0f;
  }

  // ---- loop-invariant weight fragments in VGPRs (2 N-tiles per wave) ----
  v16bf wih0[2], whh0[2][2], wih1[2][2], whh1[2][2];
  float bb0[2], bb1[2];
#pragma unroll
  for (int i = 0; i < 2; i++) {
    int n0 = wave * 32 + i * 16;
    wih0[i] = load_wfragT(Wih0, 16, 16, n0, lane, 0);
#pragma unroll
    for (int kc = 0; kc < 2; kc++) {
      whh0[i][kc] = load_wfragT(Whh0, 50, 50, n0, lane, kc * 32);
      wih1[i][kc] = load_wfragT(Wih1, 50, 50, n0, lane, kc * 32);
      whh1[i][kc] = load_wfragT(Whh1, 50, 50, n0, lane, kc * 32);
    }
    int nn = n0 + ln, g = nn >> 6, j = nn & 63;
    bb0[i] = (j < 50) ? b0[g * 50 + j] : 0.0f;
    bb1[i] = (j < 50) ? b1[g * 50 + j] : 0.0f;
  }

  // x staging: thread (m = tid>>4, f = tid&15) -> fragment slot
  const int xm = tid >> 4, xf = tid & 15;
  const int xkh = xf >> 3, xe = xf & 7;          // inverse A-map for k=f (<16)
  const float* xrow = x + (size_t)(b0r + xm) * T * F + xf;
  __bf16* const xslot = &xsf[xkh * 16 + xm][xe];

  // pre-stage x_0
  *xslot = (__bf16)xrow[0];
  __builtin_prefetch(&xrow[F], 0, 0);
  __syncthreads();

#pragma unroll 1
  for (int t = 0; t < T; t++) {
    // ---- layer-0 gates: b0 + x_t @ Wih0^T + h0 @ Whh0^T ----
    {
      v16bf ax  = *(const v16bf*)(&xsf[lane][0]);
      v16bf a0l = *(const v16bf*)(&h0frag[0][lane][0]);
      v16bf a0h = *(const v16bf*)(&h0frag[1][lane][0]);
#pragma unroll
      for (int i = 0; i < 2; i++) {
        int n0 = wave * 32 + i * 16;
        v8f acc;
#pragma unroll
        for (int r = 0; r < 8; r++) acc[r] = bb0[i];
        acc = WMMA_BF16(ax,  wih0[i],    acc);
        acc = WMMA_BF16(a0l, whh0[i][0], acc);
        acc = WMMA_BF16(a0h, whh0[i][1], acc);
#pragma unroll
        for (int r = 0; r < 8; r++) gatesS[r + 8 * khl][n0 + ln] = acc[r];
      }
    }
    __syncthreads();

    // ---- layer-0 elementwise: c,h update; scatter h into fragment order ----
#pragma unroll
    for (int u = 0; u < 4; u++) {
      int idx = tid + 256 * u, m = idx >> 6, j = idx & 63;
      float ig = gatesS[m][j],       fg = gatesS[m][64 + j];
      float gg = gatesS[m][128 + j], og = gatesS[m][192 + j];
      float c = sigf(fg) * c0s[m][j] + sigf(ig) * tanh_fast(gg);
      c0s[m][j] = c;
      float h = sigf(og) * tanh_fast(c);
      int ch = j >> 5, kp = j & 31, kh2, e2;
      inv_ak(kp, kh2, e2);
      h0frag[ch][kh2 * 16 + m][e2] = (__bf16)h;
    }
    __syncthreads();

    // ---- layer-1 gates: b1 + h0_t @ Wih1^T + h1 @ Whh1^T ----
    {
      v16bf a0l = *(const v16bf*)(&h0frag[0][lane][0]);
      v16bf a0h = *(const v16bf*)(&h0frag[1][lane][0]);
      v16bf a1l = *(const v16bf*)(&h1frag[0][lane][0]);
      v16bf a1h = *(const v16bf*)(&h1frag[1][lane][0]);
#pragma unroll
      for (int i = 0; i < 2; i++) {
        int n0 = wave * 32 + i * 16;
        v8f acc;
#pragma unroll
        for (int r = 0; r < 8; r++) acc[r] = bb1[i];
        acc = WMMA_BF16(a0l, wih1[i][0], acc);
        acc = WMMA_BF16(a0h, wih1[i][1], acc);
        acc = WMMA_BF16(a1l, whh1[i][0], acc);
        acc = WMMA_BF16(a1h, whh1[i][1], acc);
#pragma unroll
        for (int r = 0; r < 8; r++) gatesS[r + 8 * khl][n0 + ln] = acc[r];
      }
    }
    __syncthreads();

    // ---- layer-1 elementwise; also stage x_{t+1} (xsf is idle here) ----
    if (t + 1 < T) {
      *xslot = (__bf16)xrow[(size_t)(t + 1) * F];
      if (t + 2 < T) __builtin_prefetch(&xrow[(size_t)(t + 2) * F], 0, 0);
    }
#pragma unroll
    for (int u = 0; u < 4; u++) {
      int idx = tid + 256 * u, m = idx >> 6, j = idx & 63;
      float ig = gatesS[m][j],       fg = gatesS[m][64 + j];
      float gg = gatesS[m][128 + j], og = gatesS[m][192 + j];
      float c = sigf(fg) * c1s[m][j] + sigf(ig) * tanh_fast(gg);
      c1s[m][j] = c;
      float h = sigf(og) * tanh_fast(c);
      int ch = j >> 5, kp = j & 31, kh2, e2;
      inv_ak(kp, kh2, e2);
      h1frag[ch][kh2 * 16 + m][e2] = (__bf16)h;
      if (t == T - 1) h1last[m][j] = h;
    }
    __syncthreads();
  }

  // ---- FC head: sigmoid(h1_last @ Wfc^T + bfc) ----
  if (tid < 16) {
    float acc = bfc[0];
#pragma unroll
    for (int j = 0; j < 50; j++) acc += h1last[tid][j] * Wfc[j];
    out[b0r + tid] = sigf(acc);
  }
}

extern "C" void kernel_launch(void* const* d_in, const int* in_sizes, int n_in,
                              void* d_out, int out_size, void* d_ws, size_t ws_size,
                              hipStream_t stream) {
  (void)in_sizes; (void)n_in; (void)d_ws; (void)ws_size; (void)out_size;
  const float* x    = (const float*)d_in[0];
  const float* Wih0 = (const float*)d_in[1];
  const float* Whh0 = (const float*)d_in[2];
  const float* b0   = (const float*)d_in[3];
  const float* Wih1 = (const float*)d_in[4];
  const float* Whh1 = (const float*)d_in[5];
  const float* b1   = (const float*)d_in[6];
  const float* Wfc  = (const float*)d_in[7];
  const float* bfc  = (const float*)d_in[8];
  float* out = (float*)d_out;

  dim3 grid(2048 / 16);   // one WG per 16-row batch tile
  dim3 block(256);        // 8 waves (wave32)
  lstm2_fused_kernel<<<grid, block, 0, stream>>>(x, Wih0, Whh0, b0, Wih1, Whh1,
                                                 b1, Wfc, bfc, out);
}
